// TutinaModel_88648124989802
// MI455X (gfx1250) — matrix operand order repaired and checked by reading, not compiled, not run on hardware
//
#include <hip/hip_runtime.h>

#define B_TOT   4096
#define T_HISTN 168
#define T_OUTN  48
#define T_FCN   49
#define NL      64     // N_LABELS
#define FCTL    16
#define FFC     8
#define HID     32
#define GATES   128    // 4*HID
#define MLP_IN  52     // HID + FCTL + 4
#define MLP_HID 64
#define TILE_B  16
#define NTHR    128

typedef __attribute__((ext_vector_type(16))) _Float16 v16h;
typedef __attribute__((ext_vector_type(8)))  float    v8f;

__device__ __forceinline__ v8f wmma_step(v16h a, v16h b, v8f c) {
  // v_wmma_f32_16x16x32_f16 : D = A(16x32 f16) * B(32x16 f16) + C(16x16 f32)
  return __builtin_amdgcn_wmma_f32_16x16x32_f16(false, a, false, b, (short)0, c,
                                                false, false);
}

// Fast activations on the TRANS pipe only: v_exp_f32 + v_rcp_f32
// (avoids the IEEE divide expansion: div_scale/rcp/fma-fixup chains).
__device__ __forceinline__ float sigm(float x) {
  return __builtin_amdgcn_rcpf(1.0f + __expf(-x));
}
__device__ __forceinline__ float tanh_fast(float x) {
  const float e = __expf(2.0f * x);
  return 1.0f - 2.0f * __builtin_amdgcn_rcpf(e + 1.0f);
}

// A fragment (16x32 f16) from row-major LDS tile X[16][LDK], K-base kb.
// ISA layout: lanes 0-15 -> M=lane, K = kb+{0..7, 16..23};
//             lanes 16-31 -> M=lane-16, K = kb+{8..15, 24..31}.
template <int LDK>
__device__ __forceinline__ v16h load_a_frag(const _Float16 (*X)[LDK], int kb,
                                            int lane) {
  const int m  = lane & 15;
  const int kh = (lane >> 4) << 3;  // 0 or 8
  v16h a;
#pragma unroll
  for (int j = 0; j < 8; ++j) {
    a[j]     = X[m][kb + kh + j];
    a[j + 8] = X[m][kb + 16 + kh + j];
  }
  return a;
}

// B fragment (32x16 f16) from row-major f32 weights W[K][ldn], tile (kb, nb).
// ISA layout: lanes 0-15 -> N=lane, K=kb+{0..7, 16..23}; lanes 16-31 ->
// N=lane-16, K=kb+{8..15, 24..31} (matches A fragment).
__device__ __forceinline__ v16h load_b_frag(const float* W, int kb, int nb,
                                            int ldn, int kmax, int lane) {
  const int n  = lane & 15;
  const int kh = (lane >> 4) << 3;  // 0 or 8
  v16h b;
#pragma unroll
  for (int v = 0; v < 16; ++v) {
    const int k = kb + kh + (v & 7) + ((v >> 3) << 4);
    b[v] = (_Float16)((k < kmax) ? W[k * ldn + nb + n] : 0.0f);
  }
  return b;
}

__global__ __launch_bounds__(NTHR) void tutina_kernel(
    const float* __restrict__ history, const float* __restrict__ control,
    const float* __restrict__ forecasts, const float* __restrict__ h_mean,
    const float* __restrict__ h_var, const float* __restrict__ c_mean,
    const float* __restrict__ c_var, const float* __restrict__ f_mean,
    const float* __restrict__ f_var, const float* __restrict__ conv_w,
    const float* __restrict__ conv_b, const float* __restrict__ lstm_k,
    const float* __restrict__ lstm_rk, const float* __restrict__ lstm_b,
    const float* __restrict__ w1, const float* __restrict__ b1,
    const float* __restrict__ w2, const float* __restrict__ b2,
    float* __restrict__ out) {
  __shared__ _Float16 sX[TILE_B][NL];           // LSTM input tile (f16)
  __shared__ float    sZ[TILE_B][GATES];        // gate preactivations
  __shared__ _Float16 sH[TILE_B][HID];          // hidden state (f16 A operand)
  __shared__ float    sLatest[TILE_B][NL];      // running "latest" labels
  __shared__ _Float16 sHid[TILE_B][MLP_HID];    // MLP hidden (f16)
  __shared__ _Float16 sM1in[TILE_B][64];        // MLP input, K padded to 64
  __shared__ _Float16 sFc[T_OUTN][TILE_B][4];   // precomputed conv features
  __shared__ _Float16 sCtlAll[T_OUTN][TILE_B][FCTL];  // all normalized ctl
  __shared__ float sHm[NL], sHinv[NL], sCm[FCTL], sCinv[FCTL];

  const int tid  = threadIdx.x;
  const int lane = tid & 31;
  const int wave = tid >> 5;
  const int b0   = blockIdx.x * TILE_B;

  // ---- normalization stats into LDS ----
  if (tid < NL) {
    sHm[tid]   = h_mean[tid];
    sHinv[tid] = 1.0f / fmaxf(sqrtf(h_var[tid]), 1e-7f);
  } else if (tid < NL + FCTL) {
    const int j = tid - NL;
    sCm[j]   = c_mean[j];
    sCinv[j] = 1.0f / fmaxf(sqrtf(c_var[j]), 1e-7f);
  }
  const float fM = f_mean[0];
  const float fI = 1.0f / fmaxf(sqrtf(f_var[0]), 1e-7f);

  // ---- init h, latest ----
#pragma unroll
  for (int k = 0; k < 4; ++k) {
    const int i = tid + k * NTHR;  // 512 = 16*32
    sH[i >> 5][i & 31] = (_Float16)0.0f;
  }
  const int rrow = tid >> 3, c0 = (tid & 7) << 3;  // this thread's fill slot
  {
    const float* src =
        history + ((size_t)(b0 + rrow) * T_HISTN + (T_HISTN - 1)) * NL + c0;
#pragma unroll
    for (int j = 0; j < 8; ++j) sLatest[rrow][c0 + j] = src[j];
  }
  __syncthreads();  // stats visible

  // per-thread cached normalization constants
  float hm8[8], hi8[8];
#pragma unroll
  for (int j = 0; j < 8; ++j) { hm8[j] = sHm[c0 + j]; hi8[j] = sHinv[c0 + j]; }
  const int mcol = wave * 16 + (lane & 15);  // this lane's MLP output column
  const int mh   = (lane >> 4) << 3;         // D-fragment row base
  const float hmc = sHm[mcol], hic = sHinv[mcol];

  // ---- precompute conv features fc[t] (48 steps) ----
  for (int idx = tid; idx < TILE_B * T_OUTN; idx += NTHR) {
    const int m = idx & 15, t = idx >> 4;
    const float* fp = forecasts + ((size_t)(b0 + m) * T_FCN + t) * FFC;
    float acc[4];
#pragma unroll
    for (int o = 0; o < 4; ++o) acc[o] = conv_b[o];
#pragma unroll
    for (int i = 0; i < FFC; ++i) {
      const float a0 = (fp[i] - fM) * fI;
      const float a1 = (fp[FFC + i] - fM) * fI;
#pragma unroll
      for (int o = 0; o < 4; ++o)
        acc[o] += a0 * conv_w[i * 4 + o] + a1 * conv_w[32 + i * 4 + o];
    }
#pragma unroll
    for (int o = 0; o < 4; ++o) sFc[t][m][o] = (_Float16)acc[o];
  }

  // ---- precompute all normalized control tiles ----
  for (int idx = tid; idx < T_OUTN * TILE_B * FCTL; idx += NTHR) {
    const int j = idx & 15, m = (idx >> 4) & 15, t = idx >> 8;
    const float v = control[((size_t)(b0 + m) * T_OUTN + t) * FCTL + j];
    sCtlAll[t][m][j] = (_Float16)((v - sCm[j]) * sCinv[j]);
  }

  // ---- weight fragments resident in VGPRs for the whole kernel ----
  v16h Bk[2][2], Brk[2], Bw1k[2], Bw2k[2];
  float lb[2], b1v, b2v;
  {
    const int gbase = wave * 32;  // this wave's 32 gate columns
#pragma unroll
    for (int tle = 0; tle < 2; ++tle) {
#pragma unroll
      for (int kc = 0; kc < 2; ++kc)
        Bk[tle][kc] =
            load_b_frag(lstm_k, kc * 32, gbase + tle * 16, GATES, NL, lane);
      Brk[tle] = load_b_frag(lstm_rk, 0, gbase + tle * 16, GATES, HID, lane);
      lb[tle]  = lstm_b[gbase + tle * 16 + (lane & 15)];
    }
#pragma unroll
    for (int kc = 0; kc < 2; ++kc) {
      Bw1k[kc] = load_b_frag(w1, kc * 32, wave * 16, MLP_HID, MLP_IN, lane);
      Bw2k[kc] = load_b_frag(w2, kc * 32, wave * 16, NL, MLP_HID, lane);
    }
    b1v = b1[mcol];
    b2v = b2[mcol];
  }
  __syncthreads();

  float creg[4] = {0.0f, 0.0f, 0.0f, 0.0f};  // cell state in registers

  // ---- LSTM matrix part: sX, sH -> sZ (bias folded into acc init) ----
  auto lstm_wmma = [&]() {
    const v16h Ax0 = load_a_frag<NL>(sX, 0, lane);
    const v16h Ax1 = load_a_frag<NL>(sX, 32, lane);
    const v16h Ah  = load_a_frag<HID>(sH, 0, lane);
    const int colb = wave * 32 + (lane & 15);
#pragma unroll
    for (int tle = 0; tle < 2; ++tle) {
      v8f acc;
#pragma unroll
      for (int r = 0; r < 8; ++r) acc[r] = lb[tle];
      acc = wmma_step(Ax0, Bk[tle][0], acc);
      acc = wmma_step(Ax1, Bk[tle][1], acc);
      acc = wmma_step(Ah, Brk[tle], acc);
#pragma unroll
      for (int r = 0; r < 8; ++r) sZ[mh + r][colb + tle * 16] = acc[r];
    }
  };

  // ---- LSTM activation; optionally also assembles the MLP input tile ----
  auto lstm_act = [&](bool fill, int tf) {
#pragma unroll
    for (int k = 0; k < 4; ++k) {
      const int idx = tid + k * NTHR;  // 512 = 16*32, exact
      const int m = idx >> 5, j = idx & 31;
      const float zi = sZ[m][j], zf = sZ[m][32 + j];
      const float zg = sZ[m][64 + j], zo = sZ[m][96 + j];
      const float c = sigm(zf) * creg[k] + sigm(zi) * tanh_fast(zg);
      creg[k] = c;
      const _Float16 h = (_Float16)(sigm(zo) * tanh_fast(c));
      sH[m][j] = h;
      if (fill) sM1in[m][j] = h;  // MLP input cols 0..31
    }
    if (fill) {  // MLP input cols 32..63 = [ctl | fc | zeros]
#pragma unroll
      for (int k = 0; k < 4; ++k) {
        const int idx = tid + k * NTHR;
        const int m = idx >> 5, c = 32 + (idx & 31);
        _Float16 v;
        if (c < HID + FCTL)  v = sCtlAll[tf][m][c - HID];
        else if (c < MLP_IN) v = sFc[tf][m][c - HID - FCTL];
        else                 v = (_Float16)0.0f;
        sM1in[m][c] = v;
      }
    }
  };

  // ---- MLP; stage 2 also emits normalized "latest" into sX for next step ---
  auto mlp_step = [&](int t, bool write_out) {
    {
      const v16h A0 = load_a_frag<64>(sM1in, 0, lane);
      const v16h A1 = load_a_frag<64>(sM1in, 32, lane);
      v8f acc;
#pragma unroll
      for (int r = 0; r < 8; ++r) acc[r] = b1v;
      acc = wmma_step(A0, Bw1k[0], acc);
      acc = wmma_step(A1, Bw1k[1], acc);
#pragma unroll
      for (int r = 0; r < 8; ++r)
        sHid[mh + r][mcol] = (_Float16)fmaxf(acc[r], 0.0f);
    }
    __syncthreads();
    {
      const v16h A0 = load_a_frag<MLP_HID>(sHid, 0, lane);
      const v16h A1 = load_a_frag<MLP_HID>(sHid, 32, lane);
      v8f acc;
#pragma unroll
      for (int r = 0; r < 8; ++r) acc[r] = b2v;
      acc = wmma_step(A0, Bw2k[0], acc);
      acc = wmma_step(A1, Bw2k[1], acc);
#pragma unroll
      for (int r = 0; r < 8; ++r) {
        const float nl = sLatest[mh + r][mcol] + acc[r];
        sLatest[mh + r][mcol] = nl;
        sX[mh + r][mcol] = (_Float16)((nl - hmc) * hic);  // next LSTM input
        if (write_out)
          out[((size_t)(b0 + mh + r) * T_OUTN + t) * NL + mcol] = nl;
      }
    }
    __syncthreads();
  };

  // ================= encoder: 168 LSTM steps, 2 barriers/step ============
  const float* hsrc = history + ((size_t)(b0 + rrow) * T_HISTN) * NL + c0;
  float xr[8];
#pragma unroll
  for (int j = 0; j < 8; ++j) xr[j] = hsrc[j];  // t = 0 preloaded

  for (int t = 0; t < T_HISTN; ++t) {
    // fill sX from pipelined registers; issue next step's loads
#pragma unroll
    for (int j = 0; j < 8; ++j)
      sX[rrow][c0 + j] = (_Float16)((xr[j] - hm8[j]) * hi8[j]);
    const int tn = (t + 1 < T_HISTN) ? t + 1 : T_HISTN - 1;
    const float* p = hsrc + (size_t)tn * NL;
    if (t + 2 < T_HISTN) __builtin_prefetch(p + NL, 0, 0);  // global_prefetch
#pragma unroll
    for (int j = 0; j < 8; ++j) xr[j] = p[j];
    __syncthreads();  // A: sX / sH ready
    lstm_wmma();
    __syncthreads();  // B: sZ ready
    lstm_act(t == T_HISTN - 1, 0);  // last step also builds MLP input (t=0)
  }
  __syncthreads();
  // encoder-end MLP: latest += mlp(h, cn[:,0], fc[:,0]); also writes sX
  mlp_step(0, false);

  // ================= decoder: 48 steps, 4 barriers/step ==================
  for (int t = 0; t < T_OUTN; ++t) {
    lstm_wmma();  // reads sX (from mlp stage 2) and sH
    __syncthreads();
    lstm_act(true, t);  // new h/c + MLP input for this step
    __syncthreads();
    mlp_step(t, true);  // updates latest, stores preds, writes next sX
  }
}

extern "C" void kernel_launch(void* const* d_in, const int* in_sizes, int n_in,
                              void* d_out, int out_size, void* d_ws,
                              size_t ws_size, hipStream_t stream) {
  tutina_kernel<<<B_TOT / TILE_B, NTHR, 0, stream>>>(
      (const float*)d_in[0],  (const float*)d_in[1],  (const float*)d_in[2],
      (const float*)d_in[3],  (const float*)d_in[4],  (const float*)d_in[5],
      (const float*)d_in[6],  (const float*)d_in[7],  (const float*)d_in[8],
      (const float*)d_in[9],  (const float*)d_in[10], (const float*)d_in[11],
      (const float*)d_in[12], (const float*)d_in[13], (const float*)d_in[14],
      (const float*)d_in[15], (const float*)d_in[16], (const float*)d_in[17],
      (float*)d_out);
}
